// MultiscaleGNN_76175539962272
// MI455X (gfx1250) — matrix operand order repaired and checked by reading, not compiled, not run on hardware
//
#include <hip/hip_runtime.h>
#include <hip/hip_bf16.h>
#include <math.h>

// ---------------------------------------------------------------------------
// MultiscaleGNN (ChebConv K={1,2}, 4 layers) for MI455X / gfx1250 (wave32)
//   out = h @ Wc + (Lhat h) @ Wt + bc   per layer (sigmoid scale-weights folded)
//   Dense path: v_wmma_f32_16x16x32_f16 ; sparse path: f32 atomics (L2-resident h)
// ---------------------------------------------------------------------------

typedef __attribute__((ext_vector_type(16))) _Float16 v16h;
typedef __attribute__((ext_vector_type(8)))  _Float16 v8h;
typedef __attribute__((ext_vector_type(8)))  float    v8f;

#define N_NODES 50000
#define HID     256
#define LN_EPS  1e-5f

__device__ __forceinline__ void atomAddF(float* p, float v) {
    __hip_atomic_fetch_add(p, v, __ATOMIC_RELAXED, __HIP_MEMORY_SCOPE_AGENT);
}

// ---------------- degree / norm precompute --------------------------------
__global__ void deg_kernel(const int* __restrict__ row, float* __restrict__ deg, int E) {
    int i = blockIdx.x * blockDim.x + threadIdx.x;
    if (i < E) atomAddF(&deg[row[i]], 1.0f);
}

__global__ void dinv_kernel(float* __restrict__ deg, int N) {
    int i = blockIdx.x * blockDim.x + threadIdx.x;
    if (i < N) {
        float d = deg[i];
        deg[i] = (d > 0.0f) ? rsqrtf(fmaxf(d, 1.0f)) : 0.0f;
    }
}

__global__ void norm_kernel(const int* __restrict__ row, const int* __restrict__ col,
                            const float* __restrict__ dinv, float* __restrict__ norm, int E) {
    int i = blockIdx.x * blockDim.x + threadIdx.x;
    if (i < E) norm[i] = -(dinv[row[i]] * dinv[col[i]]);
}

// ---------------- f32 -> f16 conversion -----------------------------------
__global__ void cvt_kernel(const float* __restrict__ in, _Float16* __restrict__ out, size_t n) {
    size_t i = (size_t)blockIdx.x * blockDim.x + threadIdx.x;
    if (i < n) out[i] = (_Float16)in[i];
}

// ---------------- sparse propagate: out[col] += norm * h[row] -------------
// feature-parallel per edge: coalesced gather + coalesced f32 atomics
template<int FI>
__global__ void propagate_kernel(const float* __restrict__ h, const int* __restrict__ row,
                                 const int* __restrict__ col, const float* __restrict__ norm,
                                 float* __restrict__ out, int E) {
    constexpr int EPB = 256 / FI;              // edges per 256-thread block
    int eL = threadIdx.x / FI;
    int f  = threadIdx.x & (FI - 1);
    int e  = blockIdx.x * EPB + eL;
    if (e >= E) return;
    int r = row[e], c = col[e];
    float w = norm[e];
    atomAddF(&out[(size_t)c * FI + f], w * h[(size_t)r * FI + f]);
}

// ---------------- fold sigmoid scale weights into combined weights --------
// WcT/WtT stored COLUMN-major (fo x fi) so WMMA B-operand loads are contiguous
__global__ void prep_weights_kernel(const float* __restrict__ W0s0, const float* __restrict__ bs0,
                                    const float* __restrict__ W0s1, const float* __restrict__ W1s1,
                                    const float* __restrict__ bs1,
                                    const float* __restrict__ sw0, const float* __restrict__ sw1,
                                    _Float16* __restrict__ WcT, _Float16* __restrict__ WtT,
                                    float* __restrict__ bc, int fi, int fo) {
    float sig0 = 1.0f / (1.0f + expf(-sw0[0]));
    float sig1 = 1.0f / (1.0f + expf(-sw1[0]));
    int i = blockIdx.x * blockDim.x + threadIdx.x;
    int total = fi * fo;
    if (i < total) {
        int k = i / fo, n = i % fo;
        float wc = 0.5f * (sig0 * W0s0[i] + sig1 * W0s1[i]);
        float wt = 0.5f * sig1 * W1s1[i];
        WcT[(size_t)n * fi + k] = (_Float16)wc;
        WtT[(size_t)n * fi + k] = (_Float16)wt;
    }
    if (i < fo) bc[i] = 0.5f * (sig0 * bs0[i] + sig1 * bs1[i]);
}

// ---------------- WMMA GEMM: O = A16 @ WcT^T + T16 @ WtT^T ----------------
// One 16x16 output tile per wave, 8 waves per block sharing the M-tile.
// A layout: lanes 0-15 = rows, halves 0..15 -> K[0..15];
// lanes 16-31 hold K[16..31].  B from column-major WcT: lane = column,
// 16 contiguous halves = K[0..15] (lanes 0-15) / K[16..31] (lanes 16-31).
template<int K>
__global__ void gemm_wmma_kernel(const _Float16* __restrict__ A, const _Float16* __restrict__ T,
                                 const _Float16* __restrict__ WcT, const _Float16* __restrict__ WtT,
                                 float* __restrict__ O, int Nf) {
    const int lane    = threadIdx.x & 31;
    const int wave    = threadIdx.x >> 5;
    const int l15     = lane & 15;
    const int halfSel = lane >> 4;
    const int m0      = blockIdx.x * 16;

    const _Float16* aBase = A + (size_t)(m0 + l15) * K + halfSel * 16;
    const _Float16* tBase = T + (size_t)(m0 + l15) * K + halfSel * 16;

    const int numNT = Nf >> 4;
    for (int nt = wave; nt < numNT; nt += 8) {
        const int n0 = nt * 16;
        const _Float16* wcB = WcT + (size_t)(n0 + l15) * K + halfSel * 16;
        const _Float16* wtB = WtT + (size_t)(n0 + l15) * K + halfSel * 16;

        v8f acc = {};
#pragma unroll
        for (int k0 = 0; k0 < K; k0 += 32) {
            v8h alo = *(const v8h*)(aBase + k0);
            v8h ahi = *(const v8h*)(aBase + k0 + 8);
            v16h a = __builtin_shufflevector(alo, ahi, 0,1,2,3,4,5,6,7,8,9,10,11,12,13,14,15);
            v16h b = *(const v16h*)(wcB + k0);
            acc = __builtin_amdgcn_wmma_f32_16x16x32_f16(false, a, false, b,
                                                         (short)0, acc, false, false);
            v8h tlo = *(const v8h*)(tBase + k0);
            v8h thi = *(const v8h*)(tBase + k0 + 8);
            v16h t = __builtin_shufflevector(tlo, thi, 0,1,2,3,4,5,6,7,8,9,10,11,12,13,14,15);
            v16h bt = *(const v16h*)(wtB + k0);
            acc = __builtin_amdgcn_wmma_f32_16x16x32_f16(false, t, false, bt,
                                                         (short)0, acc, false, false);
        }
        // C/D layout: VGPR v -> row m0 + halfSel*8 + v, col n0 + l15
        float* op = O + (size_t)(m0 + halfSel * 8) * Nf + (n0 + l15);
#pragma unroll
        for (int v = 0; v < 8; ++v) op[(size_t)v * Nf] = acc[v];
    }
}

// ---------------- bias + (LayerNorm) + ELU, one wave32 per row ------------
__global__ void act_ln_kernel(const float* __restrict__ O, const float* __restrict__ bc,
                              const float* __restrict__ g, const float* __restrict__ bln,
                              float* __restrict__ h32, _Float16* __restrict__ h16,
                              int Nn, int fo, int doLN) {
    int gwave = (int)(((size_t)blockIdx.x * blockDim.x + threadIdx.x) >> 5);
    int lane  = threadIdx.x & 31;
    if (gwave >= Nn) return;
    const float* rowp = O + (size_t)gwave * fo;
    const int per = fo >> 5;                     // 4 or 8 elements per lane
    float vals[8];
#pragma unroll
    for (int j = 0; j < 8; ++j) {
        if (j < per) vals[j] = rowp[lane + (j << 5)] + bc[lane + (j << 5)];
        else vals[j] = 0.0f;
    }
    if (doLN) {
        float s = 0.0f;
        for (int j = 0; j < per; ++j) s += vals[j];
        for (int m = 16; m >= 1; m >>= 1) s += __shfl_xor(s, m, 32);
        float mu = s / (float)fo;
        float v = 0.0f;
        for (int j = 0; j < per; ++j) { float d = vals[j] - mu; v += d * d; }
        for (int m = 16; m >= 1; m >>= 1) v += __shfl_xor(v, m, 32);
        float r = rsqrtf(v / (float)fo + LN_EPS);
        for (int j = 0; j < per; ++j) {
            int c = lane + (j << 5);
            vals[j] = (vals[j] - mu) * r * g[c] + bln[c];
        }
    }
    for (int j = 0; j < per; ++j) {
        int c = lane + (j << 5);
        float y = vals[j] > 0.0f ? vals[j] : expm1f(vals[j]);   // ELU(alpha=1)
        h32[(size_t)gwave * fo + c] = y;
        if (h16) h16[(size_t)gwave * fo + c] = (_Float16)y;
    }
}

// ===========================================================================
extern "C" void kernel_launch(void* const* d_in, const int* in_sizes, int n_in,
                              void* d_out, int out_size, void* d_ws, size_t ws_size,
                              hipStream_t stream) {
    const int N = N_NODES;
    const int E = in_sizes[1] / 2;

    const float* x  = (const float*)d_in[0];
    const int*   ei = (const int*)d_in[1];
    const int* row = ei;          // source
    const int* col = ei + E;      // target

    // input indices (insertion order): x, edge_index,
    //   params[4 layers][ s0:{W,b}, s1:{W0,W1,b} ] -> 5 per layer starting at 2,
    //   ln_params[3]{g,b} starting at 22, scale_weights[4][2] starting at 28
    static const int FI[4] = {128, 256, 256, 256};
    static const int FO[4] = {256, 256, 256, 128};

    // workspace layout
    size_t off = 0;
    auto carve = [&](size_t bytes) {
        void* p = (char*)d_ws + off;
        off = (off + bytes + 255) & ~(size_t)255;
        return p;
    };
    float*     dinv   = (float*)carve((size_t)N * 4);
    float*     nrm    = (float*)carve((size_t)E * 4);
    float*     bufA32 = (float*)carve((size_t)N * HID * 4);   // h (f32)
    float*     bufB32 = (float*)carve((size_t)N * HID * 4);   // T1 (f32) then GEMM out
    _Float16*  bufA16 = (_Float16*)carve((size_t)N * HID * 2); // h (f16)
    _Float16*  bufB16 = (_Float16*)carve((size_t)N * HID * 2); // T1 (f16)
    _Float16*  WcT    = (_Float16*)carve((size_t)HID * HID * 2);
    _Float16*  WtT    = (_Float16*)carve((size_t)HID * HID * 2);
    float*     bc     = (float*)carve((size_t)HID * 4);

    // ---- degree / edge norm ----
    hipMemsetAsync(dinv, 0, (size_t)N * 4, stream);
    deg_kernel<<<(E + 255) / 256, 256, 0, stream>>>(row, dinv, E);
    dinv_kernel<<<(N + 255) / 256, 256, 0, stream>>>(dinv, N);
    norm_kernel<<<(E + 255) / 256, 256, 0, stream>>>(row, col, dinv, nrm, E);

    // ---- layer 0 input in f16 ----
    {
        size_t n0 = (size_t)N * FI[0];
        cvt_kernel<<<(unsigned)((n0 + 255) / 256), 256, 0, stream>>>(x, bufA16, n0);
    }

    const float* h32 = x;   // f32 h for the sparse gather
    for (int li = 0; li < 4; ++li) {
        const int fi = FI[li], fo = FO[li];
        const int base = 2 + li * 5;
        const float* W0s0 = (const float*)d_in[base + 0];
        const float* bs0  = (const float*)d_in[base + 1];
        const float* W0s1 = (const float*)d_in[base + 2];
        const float* W1s1 = (const float*)d_in[base + 3];
        const float* bs1  = (const float*)d_in[base + 4];
        const float* sw0  = (const float*)d_in[28 + li * 2];
        const float* sw1  = (const float*)d_in[29 + li * 2];

        // T1 = Lhat @ h   (scatter-add, f32 atomics; h is L2-resident: 51 MB < 192 MB)
        hipMemsetAsync(bufB32, 0, (size_t)N * fi * 4, stream);
        if (fi == 128) {
            propagate_kernel<128><<<(E + 1) / 2, 256, 0, stream>>>(h32, row, col, nrm, bufB32, E);
        } else {
            propagate_kernel<256><<<E, 256, 0, stream>>>(h32, row, col, nrm, bufB32, E);
        }
        {
            size_t nt = (size_t)N * fi;
            cvt_kernel<<<(unsigned)((nt + 255) / 256), 256, 0, stream>>>(bufB32, bufB16, nt);
        }

        // fold sigmoid scale weights -> Wc, Wt, bc (transposed f16)
        prep_weights_kernel<<<(fi * fo + 255) / 256, 256, 0, stream>>>(
            W0s0, bs0, W0s1, W1s1, bs1, sw0, sw1, WcT, WtT, bc, fi, fo);

        // O = h @ Wc + T1 @ Wt   (WMMA f16 -> f32 acc); overwrites bufB32
        if (fi == 128) {
            gemm_wmma_kernel<128><<<N / 16, 256, 0, stream>>>(bufA16, bufB16, WcT, WtT, bufB32, fo);
        } else {
            gemm_wmma_kernel<256><<<N / 16, 256, 0, stream>>>(bufA16, bufB16, WcT, WtT, bufB32, fo);
        }

        // bias + LN (layers 0..2) + ELU; write next h (f32 + f16)
        const int nblocks = (N * 32 + 255) / 256;
        if (li < 3) {
            const float* g  = (const float*)d_in[22 + li * 2];
            const float* bl = (const float*)d_in[23 + li * 2];
            act_ln_kernel<<<nblocks, 256, 0, stream>>>(bufB32, bc, g, bl,
                                                       bufA32, bufA16, N, fo, 1);
            h32 = bufA32;
        } else {
            act_ln_kernel<<<nblocks, 256, 0, stream>>>(bufB32, bc, nullptr, nullptr,
                                                       (float*)d_out, nullptr, N, fo, 0);
        }
    }
}
